// SDPAModel_36163624632549
// MI455X (gfx1250) — matrix operand order repaired and checked by reading, not compiled, not run on hardware
//
#include <hip/hip_runtime.h>

typedef _Float16 f16;
typedef __attribute__((ext_vector_type(16))) _Float16 v16h;
typedef __attribute__((ext_vector_type(8)))  _Float16 v8h;
typedef __attribute__((ext_vector_type(8)))  float    v8f;
typedef __attribute__((ext_vector_type(4)))  int      v4i;

constexpr int kB = 8, kS = 2048, kHid = 256, kHQ = 4, kHKV = 2, kD = 64, kCap = 4096;

#define AS1_ __attribute__((address_space(1)))
#define AS3_ __attribute__((address_space(3)))

#if defined(__has_builtin)
#if __has_builtin(__builtin_amdgcn_global_load_async_to_lds_b128)
#define HAVE_ASYNC_LDS 1
#endif
#endif

static __device__ __forceinline__ v16h pack16(v8h lo, v8h hi) {
  v16h r;
#pragma unroll
  for (int i = 0; i < 8; ++i) { r[i] = lo[i]; r[i + 8] = hi[i]; }
  return r;
}

static __device__ __forceinline__ v8f wmma_f16(v16h a, v16h b, v8f c) {
  // D = A(16x32 f16) x B(32x16 f16) + C(16x16 f32)
  return __builtin_amdgcn_wmma_f32_16x16x32_f16(false, a, false, b, (short)0, c,
                                                false, false);
}

static __device__ __forceinline__ v8f vzero() {
  v8f z = {0.f, 0.f, 0.f, 0.f, 0.f, 0.f, 0.f, 0.f};
  return z;
}

// ---------------------------------------------------------------------------
// Kernel A: qkv = x @ Wqkv^T + b, RoPE on q/k, scatter to q_ws and kv cache.
// Grid: (B*S)/16 blocks of 256 threads (8 waves). Wave w owns N cols [64w,64w+64).
// ---------------------------------------------------------------------------
__global__ __launch_bounds__(256) void qkv_rope_kernel(
    const f16* __restrict__ x, const f16* __restrict__ Wqkv,
    const f16* __restrict__ bqkv, const float* __restrict__ rope,
    const int* __restrict__ ctxp, f16* __restrict__ q_ws,
    f16* __restrict__ kv_out) {
  const int past_len = *ctxp - kS;
  const int lane = threadIdx.x & 31;
  const int wave = threadIdx.x >> 5;
  const int l15 = lane & 15;
  const int lh = lane >> 4;
  const int mBase = blockIdx.x << 4;
  const int n0 = wave << 6;

  v8f acc[4] = {vzero(), vzero(), vzero(), vzero()};

  const f16* xrow = x + (size_t)(mBase + l15) * kHid;
#pragma unroll
  for (int kk = 0; kk < 8; ++kk) {
    const int c0 = kk * 32 + 16 * lh;
    v16h a = pack16(*(const v8h*)(xrow + c0), *(const v8h*)(xrow + c0 + 8));
#pragma unroll
    for (int t = 0; t < 4; ++t) {
      const int n = n0 + t * 16 + l15;
      v16h b = *(const v16h*)(Wqkv + (size_t)n * kHid + kk * 32 + 16 * lh);
      acc[t] = wmma_f16(a, b, acc[t]);
    }
  }

#pragma unroll
  for (int t = 0; t < 4; ++t) {
    const float bias = (float)bqkv[n0 + t * 16 + l15];
#pragma unroll
    for (int r = 0; r < 8; ++r) acc[t][r] += bias;
  }

  const int rowA = mBase + 8 * lh;

  if (n0 < 256) {  // Q columns: RoPE, write to q workspace [b][hq][s][d]
    const int h = n0 >> 6;
#pragma unroll
    for (int t = 0; t < 2; ++t) {
      const int d1 = t * 16 + l15;
#pragma unroll
      for (int r = 0; r < 8; ++r) {
        const int gm = rowA + r;
        const int bb = gm >> 11;
        const int s = gm & (kS - 1);
        const int pos = past_len + s;
        const float cs = rope[pos * kD + d1];
        const float sn = rope[pos * kD + 32 + d1];
        const float x1 = acc[t][r], x2 = acc[t + 2][r];
        f16* qp = q_ws + (((size_t)bb * kHQ + h) * kS + s) * kD;
        qp[d1] = (f16)(x1 * cs - x2 * sn);
        qp[d1 + 32] = (f16)(x1 * sn + x2 * cs);
      }
    }
  } else if (n0 < 384) {  // K columns: RoPE, write into kv cache [b][0][h][pos][d]
    const int h = (n0 - 256) >> 6;
#pragma unroll
    for (int t = 0; t < 2; ++t) {
      const int d1 = t * 16 + l15;
#pragma unroll
      for (int r = 0; r < 8; ++r) {
        const int gm = rowA + r;
        const int bb = gm >> 11;
        const int s = gm & (kS - 1);
        const int pos = past_len + s;
        const float cs = rope[pos * kD + d1];
        const float sn = rope[pos * kD + 32 + d1];
        const float x1 = acc[t][r], x2 = acc[t + 2][r];
        f16* kp = kv_out + ((((size_t)bb * 2 + 0) * kHKV + h) * kCap + pos) * kD;
        kp[d1] = (f16)(x1 * cs - x2 * sn);
        kp[d1 + 32] = (f16)(x1 * sn + x2 * cs);
      }
    }
  } else {  // V columns: write into kv cache [b][1][h][pos][d]
    const int h = (n0 - 384) >> 6;
#pragma unroll
    for (int t = 0; t < 4; ++t) {
      const int d = t * 16 + l15;
#pragma unroll
      for (int r = 0; r < 8; ++r) {
        const int gm = rowA + r;
        const int bb = gm >> 11;
        const int s = gm & (kS - 1);
        const int pos = past_len + s;
        f16* vp = kv_out + ((((size_t)bb * 2 + 1) * kHKV + h) * kCap + pos) * kD;
        vp[d] = (f16)acc[t][r];
      }
    }
  }
}

// ---------------------------------------------------------------------------
// Kernel B: causal flash attention with WMMA.
// Block = 256 threads (8 waves) = 128 query rows of one (b, hq). Grid = B*HQ*S/128.
// ---------------------------------------------------------------------------
__global__ __launch_bounds__(256) void attn_kernel(const f16* __restrict__ q_ws,
                                                   const f16* __restrict__ kv,
                                                   const int* __restrict__ ctxp,
                                                   f16* __restrict__ attn_ws) {
  __shared__ __align__(64) f16 Klds[32][64];    // [t][d]
  __shared__ __align__(64) f16 VTlds[64][48];   // [d][t], padded stride 48
  __shared__ __align__(64) f16 Plds[8][16][32]; // per-wave probs [s][t]

  const int ctx = *ctxp;
  const int tid = threadIdx.x;
  const int lane = tid & 31;
  const int wave = tid >> 5;
  const int l15 = lane & 15;
  const int lh = lane >> 4;

  const int wg = blockIdx.x;
  const int qb = wg & 15;        // S/128 = 16 query blocks
  const int hq = (wg >> 4) & 3;  // query head
  const int b = wg >> 6;
  const int hk = hq >> 1;        // G = 2

  const int s0 = qb << 7;
  const int sW = s0 + (wave << 4);  // wave's first query row

  const f16* qbase = q_ws + (((size_t)b * kHQ + hq) * kS + sW) * kD;
  const f16* kbase = kv + (((size_t)b * 2 + 0) * kHKV + hk) * (size_t)kCap * kD;
  const f16* vbase = kv + (((size_t)b * 2 + 1) * kHKV + hk) * (size_t)kCap * kD;

  // preload Q A-fragments for both K-steps (kept in VGPRs for the whole loop)
  v16h qf[2];
  {
    const f16* qrow = qbase + (size_t)l15 * kD;
#pragma unroll
    for (int kk = 0; kk < 2; ++kk) {
      const int c0 = kk * 32 + 16 * lh;
      qf[kk] = pack16(*(const v8h*)(qrow + c0), *(const v8h*)(qrow + c0 + 8));
    }
  }

  v8f oacc[4] = {vzero(), vzero(), vzero(), vzero()};
  float mrow[8], lrow[8];
#pragma unroll
  for (int r = 0; r < 8; ++r) { mrow[r] = -__builtin_inff(); lrow[r] = 0.f; }

  const int tEnd = (ctx < s0 + 128) ? ctx : (s0 + 128);
  const int sMaxW = sW + 15;
  const float scale = 0.125f;  // 1/sqrt(64)

  const int tr = tid >> 3;         // 0..31 kv row for staging
  const int dc = (tid & 7) << 3;   // 0..56 d offset

  for (int t0 = 0; t0 < tEnd; t0 += 32) {
    // cooperative staging: K as-is (async DMA to LDS when available), V transposed
    {
      const f16* kgp = kbase + (size_t)(t0 + tr) * kD + dc;
#if defined(HAVE_ASYNC_LDS)
      __builtin_amdgcn_global_load_async_to_lds_b128(
          (AS1_ v4i*)(unsigned long long)kgp,
          (AS3_ v4i*)(unsigned long long)&Klds[tr][dc], 0, 0);
#else
      v8h k8 = *(const v8h*)kgp;
      *(v8h*)(&Klds[tr][dc]) = k8;
#endif
      v8h v8v = *(const v8h*)(vbase + (size_t)(t0 + tr) * kD + dc);
#pragma unroll
      for (int i = 0; i < 8; ++i) VTlds[dc + i][tr] = v8v[i];
      if (t0 + 32 < tEnd) {
        __builtin_prefetch(kbase + (size_t)(t0 + 32 + tr) * kD + dc, 0, 1);
        __builtin_prefetch(vbase + (size_t)(t0 + 32 + tr) * kD + dc, 0, 1);
      }
#if defined(HAVE_ASYNC_LDS)
#if __has_builtin(__builtin_amdgcn_s_wait_asynccnt)
      __builtin_amdgcn_s_wait_asynccnt(0);
#else
      asm volatile("s_wait_asynccnt 0" ::: "memory");
#endif
#endif
    }
    __syncthreads();

    if (t0 <= sMaxW) {  // wave-uniform: EXEC stays all-ones for WMMA
      // S = Q K^T over D=64 (2 K-steps), two 16-wide t tiles
      v8f sacc[2] = {vzero(), vzero()};
#pragma unroll
      for (int tt = 0; tt < 2; ++tt) {
#pragma unroll
        for (int kk = 0; kk < 2; ++kk) {
          v16h bf = *(const v16h*)(&Klds[tt * 16 + l15][kk * 32 + 16 * lh]);
          sacc[tt] = wmma_f16(qf[kk], bf, sacc[tt]);
        }
      }
      const int tg0 = t0 + l15;
      const int tg1 = t0 + 16 + l15;
#pragma unroll
      for (int r = 0; r < 8; ++r) {
        const int sg = sW + r + 8 * lh;
        float v0 = sacc[0][r] * scale;
        float v1 = sacc[1][r] * scale;
        if (tg0 > sg) v0 = -3.0e38f;  // causal mask j <= i
        if (tg1 > sg) v1 = -3.0e38f;
        float vm = fmaxf(v0, v1);
        vm = fmaxf(vm, __shfl_xor(vm, 8, 16));
        vm = fmaxf(vm, __shfl_xor(vm, 4, 16));
        vm = fmaxf(vm, __shfl_xor(vm, 2, 16));
        vm = fmaxf(vm, __shfl_xor(vm, 1, 16));
        const float mnew = fmaxf(mrow[r], vm);
        const float alpha = __expf(mrow[r] - mnew);
        mrow[r] = mnew;
        const float p0 = __expf(v0 - mnew);
        const float p1 = __expf(v1 - mnew);
        float ps = p0 + p1;
        ps += __shfl_xor(ps, 8, 16);
        ps += __shfl_xor(ps, 4, 16);
        ps += __shfl_xor(ps, 2, 16);
        ps += __shfl_xor(ps, 1, 16);
        lrow[r] = lrow[r] * alpha + ps;
#pragma unroll
        for (int dt = 0; dt < 4; ++dt) oacc[dt][r] *= alpha;
        Plds[wave][r + 8 * lh][l15] = (f16)p0;
        Plds[wave][r + 8 * lh][16 + l15] = (f16)p1;
      }
      // per-wave LDS round-trip: C-layout probs -> A-layout fragment
      asm volatile("s_wait_dscnt 0" ::: "memory");
      const f16* prow = &Plds[wave][l15][0];
      const int c0 = 16 * lh;
      v16h pa = pack16(*(const v8h*)(prow + c0), *(const v8h*)(prow + c0 + 8));
#pragma unroll
      for (int dt = 0; dt < 4; ++dt) {
        v16h vb = *(const v16h*)(&VTlds[dt * 16 + l15][16 * lh]);
        oacc[dt] = wmma_f16(pa, vb, oacc[dt]);
      }
    }
    __syncthreads();
  }

  // normalize and write attention output [b][s][hq*64 + d]
#pragma unroll
  for (int r = 0; r < 8; ++r) {
    const float inv = 1.0f / lrow[r];
    const int sg = sW + r + 8 * lh;
    f16* orow = attn_ws + ((size_t)b * kS + sg) * kHid + hq * kD;
#pragma unroll
    for (int dt = 0; dt < 4; ++dt)
      orow[dt * 16 + l15] = (f16)(oacc[dt][r] * inv);
  }
}

// ---------------------------------------------------------------------------
// Kernel C: out = attn @ Wout^T. Grid (B*S)/16 blocks of 128 threads (4 waves).
// ---------------------------------------------------------------------------
__global__ __launch_bounds__(128) void outproj_kernel(
    const f16* __restrict__ attn, const f16* __restrict__ Wout,
    f16* __restrict__ out) {
  const int lane = threadIdx.x & 31;
  const int wave = threadIdx.x >> 5;
  const int l15 = lane & 15;
  const int lh = lane >> 4;
  const int mBase = blockIdx.x << 4;
  const int n0 = wave << 6;

  v8f acc[4] = {vzero(), vzero(), vzero(), vzero()};
  const f16* arow = attn + (size_t)(mBase + l15) * kHid;
#pragma unroll
  for (int kk = 0; kk < 8; ++kk) {
    const int c0 = kk * 32 + 16 * lh;
    v16h a = pack16(*(const v8h*)(arow + c0), *(const v8h*)(arow + c0 + 8));
#pragma unroll
    for (int t = 0; t < 4; ++t) {
      const int n = n0 + t * 16 + l15;
      v16h b = *(const v16h*)(Wout + (size_t)n * kHid + kk * 32 + 16 * lh);
      acc[t] = wmma_f16(a, b, acc[t]);
    }
  }
#pragma unroll
  for (int t = 0; t < 4; ++t) {
    const int n = n0 + t * 16 + l15;
#pragma unroll
    for (int r = 0; r < 8; ++r) {
      const int gm = mBase + r + 8 * lh;
      out[(size_t)gm * kHid + n] = (f16)acc[t][r];
    }
  }
}

// ---------------------------------------------------------------------------
extern "C" void kernel_launch(void* const* d_in, const int* in_sizes, int n_in,
                              void* d_out, int out_size, void* d_ws,
                              size_t ws_size, hipStream_t stream) {
  (void)in_sizes; (void)n_in; (void)out_size; (void)ws_size;
  const f16* x = (const f16*)d_in[0];
  const f16* kv_in = (const f16*)d_in[1];
  const float* rope = (const float*)d_in[2];
  const f16* Wqkv = (const f16*)d_in[3];
  const f16* bqkv = (const f16*)d_in[4];
  const f16* Wout = (const f16*)d_in[5];
  const int* ctxp = (const int*)d_in[6];

  f16* out = (f16*)d_out;                                  // [B][S][HID]
  f16* kv_out = out + (size_t)kB * kS * kHid;              // [B][2][HKV][CAP][D]
  f16* q_ws = (f16*)d_ws;                                  // [B][HQ][S][D]
  f16* attn_ws = q_ws + (size_t)kB * kHQ * kS * kD;        // [B][S][HID]

  // pass through untouched cache regions, then overwrite active slice
  (void)hipMemcpyAsync(kv_out, kv_in,
                       (size_t)kB * 2 * kHKV * kCap * kD * sizeof(f16),
                       hipMemcpyDeviceToDevice, stream);

  qkv_rope_kernel<<<(kB * kS) / 16, 256, 0, stream>>>(x, Wqkv, bqkv, rope, ctxp,
                                                      q_ws, kv_out);
  attn_kernel<<<kB * kHQ * (kS / 128), 256, 0, stream>>>(q_ws, kv_out, ctxp,
                                                         attn_ws);
  outproj_kernel<<<(kB * kS) / 16, 128, 0, stream>>>(attn_ws, Wout, out);
}
